// EncoderLayer_29772713296087
// MI455X (gfx1250) — matrix operand order repaired and checked by reading, not compiled, not run on hardware
//
#include <hip/hip_runtime.h>
#include <hip/hip_bf16.h>
#include <stdint.h>

typedef __bf16 bf16;
typedef __attribute__((ext_vector_type(8)))  __bf16 v8bf;
typedef __attribute__((ext_vector_type(16))) __bf16 v16bf;
typedef __attribute__((ext_vector_type(8)))  float  v8f;
typedef __attribute__((ext_vector_type(4))) unsigned int u32x4;
typedef __attribute__((ext_vector_type(8))) int i32x8;
typedef __attribute__((ext_vector_type(4))) int i32x4;

#define D_MODEL 1024
#define NUM_HEADS 16
#define D_FF 4096
#define BATCH 8
#define SEQ 1024
#define MTOK (BATCH * SEQ)   // 8192 tokens

#define HAVE_TDM (__has_builtin(__builtin_amdgcn_tensor_load_to_lds) && \
                  __has_builtin(__builtin_amdgcn_s_wait_tensorcnt))

__device__ __forceinline__ v8bf ld8(const bf16* p) { return *(const v8bf*)p; }

__device__ __forceinline__ v16bf make_frag(v8bf lo, v8bf hi) {
  return __builtin_shufflevector(lo, hi, 0,1,2,3,4,5,6,7,8,9,10,11,12,13,14,15);
}

__device__ __forceinline__ v8f wmma_bf16(v16bf a, v16bf b, v8f c) {
  return __builtin_amdgcn_wmma_f32_16x16x32_bf16(false, a, false, b, (short)0, c, false, false);
}

#if HAVE_TDM
// ---------------------------------------------------------------------------
// Tensor Data Mover: DMA one 2D tile (bf16 elements) global -> LDS.
// D# built per CDNA5 ISA 8.3/8.4: count=1, type=2(image), data_size=1(2B),
// pad_enable so the LDS destination keeps our padded row stride.
//   pad_interval_code: 3 -> pad every 16 DWORDs (64B row), 4 -> every 32 DW.
//   pad_amount_code:   3 -> insert 4 DWORDs (16B) of padding.
// ---------------------------------------------------------------------------
__device__ __forceinline__ void tdm_load_tile_2d(
    unsigned lds_off, const void* tile_base,
    unsigned tensor_d0, unsigned tensor_d1, unsigned stride0_elems,
    unsigned tile_d0, unsigned tile_d1,
    unsigned pad_interval_code, unsigned pad_amount_code) {
  unsigned long long ga = (unsigned long long)(uintptr_t)tile_base;
  u32x4 g0;
  g0.x = 1u;                                   // count=1 (valid), user mode
  g0.y = lds_off;                              // lds_addr (bytes)
  g0.z = (unsigned)ga;                         // global_addr[31:0]
  g0.w = (unsigned)(ga >> 32) | (2u << 30);    // global_addr[56:32] | type=2
  i32x8 g1;
  g1[0] = (int)((1u << 16)                     // data_size = 2 bytes
              | (1u << 20)                     // pad_enable
              | (pad_interval_code << 22)
              | (pad_amount_code << 25));
  g1[1] = (int)((tensor_d0 & 0xFFFFu) << 16);                       // dim0[15:0]
  g1[2] = (int)(((tensor_d0 >> 16) & 0xFFFFu) |
                ((tensor_d1 & 0xFFFFu) << 16));                     // dim0[31:16]|dim1[15:0]
  g1[3] = (int)(((tensor_d1 >> 16) & 0xFFFFu) | (tile_d0 << 16));   // dim1[31:16]|tile0
  g1[4] = (int)(tile_d1 & 0xFFFFu);                                 // tile1 (tile2=0)
  g1[5] = (int)stride0_elems;                                       // dim0_stride[31:0]
  g1[6] = 0;                                                        // stride hi / dim1_stride
  g1[7] = 0;
  const i32x4 z4 = {0, 0, 0, 0};
#if __clang_major__ >= 23
  const i32x8 z8 = {0, 0, 0, 0, 0, 0, 0, 0};
  __builtin_amdgcn_tensor_load_to_lds(g0, g1, z4, z4, z8, 0);
#else
  __builtin_amdgcn_tensor_load_to_lds(g0, g1, z4, z4, 0);
#endif
}
#endif  // HAVE_TDM

// ---------------------------------------------------------------------------
// fp32 [R][C] -> bf16 transposed [C][R]
// ---------------------------------------------------------------------------
__global__ __launch_bounds__(256) void transpose_f32_to_bf16(
    const float* __restrict__ src, bf16* __restrict__ dst, int R, int C) {
  __shared__ float tile[32][33];
  const int tx = threadIdx.x, ty = threadIdx.y;       // 32 x 8
  const int x = blockIdx.x * 32 + tx;
  const int y0 = blockIdx.y * 32;
  #pragma unroll
  for (int j = 0; j < 4; ++j)
    tile[ty + j * 8][tx] = src[(size_t)(y0 + ty + j * 8) * C + x];
  __syncthreads();
  const int xo = blockIdx.y * 32 + tx;
  const int yo0 = blockIdx.x * 32;
  #pragma unroll
  for (int j = 0; j < 4; ++j)
    dst[(size_t)(yo0 + ty + j * 8) * R + xo] = (bf16)tile[tx][ty + j * 8];
}

// ---------------------------------------------------------------------------
// LayerNorm over D=1024, one row per 256-thread block, bf16 output
// ---------------------------------------------------------------------------
__global__ __launch_bounds__(256) void ln_bf16_kernel(
    const float* __restrict__ x, const float* __restrict__ gma,
    const float* __restrict__ bta, bf16* __restrict__ out) {
  __shared__ float red[16];
  const int row = blockIdx.x, tid = threadIdx.x;
  const float4 v = ((const float4*)(x + (size_t)row * D_MODEL))[tid];
  float s  = v.x + v.y + v.z + v.w;
  float s2 = v.x * v.x + v.y * v.y + v.z * v.z + v.w * v.w;
  s  += __shfl_xor(s, 1);  s2 += __shfl_xor(s2, 1);
  s  += __shfl_xor(s, 2);  s2 += __shfl_xor(s2, 2);
  s  += __shfl_xor(s, 4);  s2 += __shfl_xor(s2, 4);
  s  += __shfl_xor(s, 8);  s2 += __shfl_xor(s2, 8);
  s  += __shfl_xor(s, 16); s2 += __shfl_xor(s2, 16);
  const int wid = tid >> 5;
  if ((tid & 31) == 0) { red[wid] = s; red[8 + wid] = s2; }
  __syncthreads();
  if (tid == 0) {
    float a = 0.f, b = 0.f;
    #pragma unroll
    for (int i = 0; i < 8; ++i) { a += red[i]; b += red[8 + i]; }
    red[0] = a; red[8] = b;
  }
  __syncthreads();
  const float mu = red[0] * (1.0f / D_MODEL);
  const float var = red[8] * (1.0f / D_MODEL) - mu * mu;
  const float rs = rsqrtf(var + 1e-6f);
  const float4 g4 = ((const float4*)gma)[tid];
  const float4 b4 = ((const float4*)bta)[tid];
  bf16* orow = out + (size_t)row * D_MODEL + tid * 4;
  orow[0] = (bf16)((v.x - mu) * rs * g4.x + b4.x);
  orow[1] = (bf16)((v.y - mu) * rs * g4.y + b4.y);
  orow[2] = (bf16)((v.z - mu) * rs * g4.z + b4.z);
  orow[3] = (bf16)((v.w - mu) * rs * g4.w + b4.w);
}

// ---------------------------------------------------------------------------
// WMMA GEMM: C[M][N] = A[M][K](bf16) * Wt[N][K]^T(bf16) + bias
//   mode 0: out = bf16(C + bias)
//   mode 1: out = bf16(relu(C + bias))
//   mode 2: outF = res + C + bias  (fp32)
// Block 256 thr (8 waves), block tile 128x128, BK=32, wave tile 32x64.
// Tiles staged by TDM (one tensor_load_to_lds per tile) when available.
// ---------------------------------------------------------------------------
#define SA 40
__global__ __launch_bounds__(256) void gemm_bf16_wmma(
    const bf16* __restrict__ A, const bf16* __restrict__ Wt,
    const float* __restrict__ bias, const float* __restrict__ res,
    float* __restrict__ outF, bf16* __restrict__ outB,
    int M, int N, int K, int mode) {
  __shared__ bf16 As[128][SA];
  __shared__ bf16 Bs[128][SA];
  const int tid = threadIdx.x;
  const int lane = tid & 31, w = tid >> 5;
  const int lr = lane & 15, lh = lane >> 4;
  const int wm = w >> 1, wn = w & 1;
  const int m0 = blockIdx.y * 128;
  const int n0 = blockIdx.x * 128;

  v8f acc[2][4];
  const v8f vzero = {0.f, 0.f, 0.f, 0.f, 0.f, 0.f, 0.f, 0.f};
  #pragma unroll
  for (int i = 0; i < 2; ++i)
    #pragma unroll
    for (int j = 0; j < 4; ++j) acc[i][j] = vzero;

  const int nkt = K >> 5;
  for (int kt = 0; kt < nkt; ++kt) {
    const int k0 = kt << 5;
    __syncthreads();   // previous tile fully consumed
#if HAVE_TDM
    if (w == 0) {
      tdm_load_tile_2d((unsigned)(uintptr_t)&As[0][0],
                       A + (size_t)m0 * K + k0,
                       (unsigned)K, (unsigned)M, (unsigned)K, 32, 128, 3, 3);
      tdm_load_tile_2d((unsigned)(uintptr_t)&Bs[0][0],
                       Wt + (size_t)n0 * K + k0,
                       (unsigned)K, (unsigned)N, (unsigned)K, 32, 128, 3, 3);
      __builtin_amdgcn_s_wait_tensorcnt(0);
    }
#else
    #pragma unroll
    for (int u = 0; u < 2; ++u) {           // A tile 128x32: 512 v8bf
      const int vv = tid + u * 256;
      const int r = vv >> 2, cv = vv & 3;
      *(v8bf*)&As[r][cv * 8] = ld8(&A[(size_t)(m0 + r) * K + k0 + cv * 8]);
    }
    #pragma unroll
    for (int u = 0; u < 2; ++u) {           // B tile 128x32: 512 v8bf
      const int vv = tid + u * 256;
      const int r = vv >> 2, cv = vv & 3;
      *(v8bf*)&Bs[r][cv * 8] = ld8(&Wt[(size_t)(n0 + r) * K + k0 + cv * 8]);
    }
    if (kt + 1 < nkt) {
      __builtin_prefetch(&A[(size_t)(m0 + (tid >> 1)) * K + k0 + 32], 0, 1);
      __builtin_prefetch(&Wt[(size_t)(n0 + (tid >> 1)) * K + k0 + 32], 0, 1);
    }
#endif
    __syncthreads();

    v16bf af[2], bfg[4];
    #pragma unroll
    for (int i = 0; i < 2; ++i) {           // A frag: row=lr, K {ka..+7, 16+ka..}
      const bf16* p = &As[wm * 32 + i * 16 + lr][lh * 8];
      af[i] = make_frag(ld8(p), ld8(p + 16));
    }
    #pragma unroll
    for (int j = 0; j < 4; ++j) {           // B frag: col=lr, K {kb..+7, 16+kb..}
      const bf16* p = &Bs[wn * 64 + j * 16 + lr][lh * 8];
      bfg[j] = make_frag(ld8(p), ld8(p + 16));
    }
    #pragma unroll
    for (int i = 0; i < 2; ++i)
      #pragma unroll
      for (int j = 0; j < 4; ++j)
        acc[i][j] = wmma_bf16(af[i], bfg[j], acc[i][j]);
  }

  #pragma unroll
  for (int i = 0; i < 2; ++i)
    #pragma unroll
    for (int j = 0; j < 4; ++j) {
      const int colb = n0 + wn * 64 + j * 16 + lr;
      const float bs = bias[colb];
      #pragma unroll
      for (int g = 0; g < 8; ++g) {
        const int row = m0 + wm * 32 + i * 16 + g + lh * 8;
        float v = acc[i][j][g] + bs;
        const size_t idx = (size_t)row * N + colb;
        if (mode == 2) {
          outF[idx] = res[idx] + v;
        } else {
          if (mode == 1 && v < 0.f) v = 0.f;
          outB[idx] = (bf16)v;
        }
      }
    }
}

// ---------------------------------------------------------------------------
// Flash attention: Q,K,V bf16 [B*L][H*64]; per block: one (b,h), 128 q rows,
// 8 waves x 16 rows. Key tiles of 32, online softmax, WMMA for QK^T and PV.
// K tile staged by TDM when available; V staged transposed by VALU.
// ---------------------------------------------------------------------------
__global__ __launch_bounds__(256) void attn_flash_wmma(
    const bf16* __restrict__ Q, const bf16* __restrict__ Kq,
    const bf16* __restrict__ V, const int* __restrict__ mask,
    bf16* __restrict__ O) {
  __shared__ bf16 Ks[32][72];      // [key][d] row-major (128B rows + 16B pad)
  __shared__ bf16 Vst[64][40];     // [d][key] transposed
  __shared__ bf16 Ps[8][16][40];   // per-wave probs re-swizzle buffer
  __shared__ int  mLds[SEQ];

  const int tid = threadIdx.x, lane = tid & 31, w = tid >> 5;
  const int lr = lane & 15, lh = lane >> 4;
  const int bh = blockIdx.y, b = bh >> 4, h = bh & 15;
  const int q0 = blockIdx.x * 128;
  const int D = D_MODEL;

  for (int i = tid; i < SEQ; i += 256) mLds[i] = mask[b * SEQ + i];

  v16bf aq[2];
  {
    const bf16* qp = Q + (size_t)(b * SEQ + q0 + w * 16 + lr) * D + h * 64;
    #pragma unroll
    for (int i = 0; i < 2; ++i) {
      const bf16* p = qp + i * 32 + lh * 8;
      aq[i] = make_frag(ld8(p), ld8(p + 16));
    }
  }

  const v8f vzero = {0.f, 0.f, 0.f, 0.f, 0.f, 0.f, 0.f, 0.f};
  v8f acc[4];
  #pragma unroll
  for (int j = 0; j < 4; ++j) acc[j] = vzero;
  float mrow[8], srow[8];
  #pragma unroll
  for (int g = 0; g < 8; ++g) { mrow[g] = -3.0e38f; srow[g] = 0.f; }

  for (int kt = 0; kt < SEQ / 32; ++kt) {
    const int kk = kt * 32;
    __syncthreads();
#if HAVE_TDM
    if (w == 0)   // K tile 32x64, rows 128B + 16B pad -> codes (4,3)
      tdm_load_tile_2d((unsigned)(uintptr_t)&Ks[0][0],
                       Kq + (size_t)(b * SEQ + kk) * D + h * 64,
                       (unsigned)D, (unsigned)(BATCH * SEQ), (unsigned)D,
                       64, 32, 4, 3);
#else
    {
      const int r = tid >> 3, cv = tid & 7;
      *(v8bf*)&Ks[r][cv * 8] =
          ld8(&Kq[(size_t)(b * SEQ + kk + r) * D + h * 64 + cv * 8]);
    }
#endif
    {   // stage V tile transposed -> Vst[d][key]
      const int key = tid >> 3, dv = tid & 7;
      v8bf vv = ld8(&V[(size_t)(b * SEQ + kk + key) * D + h * 64 + dv * 8]);
      #pragma unroll
      for (int z = 0; z < 8; ++z) Vst[dv * 8 + z][key] = vv[z];
    }
#if HAVE_TDM
    if (w == 0) __builtin_amdgcn_s_wait_tensorcnt(0);
#endif
    __syncthreads();

    float sf[2][8];
    #pragma unroll
    for (int kt2 = 0; kt2 < 2; ++kt2) {
      v8f s = vzero;
      #pragma unroll
      for (int i = 0; i < 2; ++i) {   // B frag of K^T: col=key(lr), K=d
        const bf16* p = &Ks[kt2 * 16 + lr][i * 32 + lh * 8];
        v16bf bk = make_frag(ld8(p), ld8(p + 16));
        s = wmma_bf16(aq[i], bk, s);
      }
      const int mv = mLds[kk + kt2 * 16 + lr];
      #pragma unroll
      for (int g = 0; g < 8; ++g) {
        float x = s[g] * 0.125f;            // 1/sqrt(64)
        if (mv == 0) x = -1.0e9f;
        sf[kt2][g] = x;
      }
    }
    // online softmax (row = g + 8*lh; reduce over lr via xor 1,2,4,8)
    #pragma unroll
    for (int g = 0; g < 8; ++g) {
      float t = fmaxf(sf[0][g], sf[1][g]);
      t = fmaxf(t, __shfl_xor(t, 1));
      t = fmaxf(t, __shfl_xor(t, 2));
      t = fmaxf(t, __shfl_xor(t, 4));
      t = fmaxf(t, __shfl_xor(t, 8));
      const float mn = fmaxf(mrow[g], t);
      const float sc = __expf(mrow[g] - mn);
      const float p0 = __expf(sf[0][g] - mn);
      const float p1 = __expf(sf[1][g] - mn);
      Ps[w][g + lh * 8][lr]      = (bf16)p0;
      Ps[w][g + lh * 8][16 + lr] = (bf16)p1;
      float pp = p0 + p1;
      pp += __shfl_xor(pp, 1);
      pp += __shfl_xor(pp, 2);
      pp += __shfl_xor(pp, 4);
      pp += __shfl_xor(pp, 8);
      srow[g] = srow[g] * sc + pp;
      mrow[g] = mn;
      #pragma unroll
      for (int j = 0; j < 4; ++j) acc[j][g] *= sc;
    }
    __syncthreads();

    // P (16x32 over keys) as A fragment
    v16bf pa;
    {
      const bf16* p = &Ps[w][lr][lh * 8];
      pa = make_frag(ld8(p), ld8(p + 16));
    }
    #pragma unroll
    for (int j = 0; j < 4; ++j) {   // V as B frag: col=dk(lr), K=key(lh*8.., 16+lh*8..)
      const bf16* p = &Vst[j * 16 + lr][lh * 8];
      v16bf vb = make_frag(ld8(p), ld8(p + 16));
      acc[j] = wmma_bf16(pa, vb, acc[j]);
    }
  }

  #pragma unroll
  for (int g = 0; g < 8; ++g) {
    const float inv = 1.0f / srow[g];
    const int row = q0 + w * 16 + g + lh * 8;
    #pragma unroll
    for (int j = 0; j < 4; ++j)
      O[(size_t)(b * SEQ + row) * D + h * 64 + j * 16 + lr] =
          (bf16)(acc[j][g] * inv);
  }
}

// ---------------------------------------------------------------------------
extern "C" void kernel_launch(void* const* d_in, const int* in_sizes, int n_in,
                              void* d_out, int out_size, void* d_ws, size_t ws_size,
                              hipStream_t stream) {
  const float* x     = (const float*)d_in[0];
  const int*   emask = (const int*)d_in[1];
  const float* ln1_g = (const float*)d_in[2];
  const float* ln1_b = (const float*)d_in[3];
  const float* Wq = (const float*)d_in[4];  const float* bq = (const float*)d_in[5];
  const float* Wk = (const float*)d_in[6];  const float* bk = (const float*)d_in[7];
  const float* Wv = (const float*)d_in[8];  const float* bv = (const float*)d_in[9];
  const float* Wo = (const float*)d_in[10]; const float* bo = (const float*)d_in[11];
  const float* ln2_g = (const float*)d_in[12];
  const float* ln2_b = (const float*)d_in[13];
  const float* W1 = (const float*)d_in[14]; const float* b1 = (const float*)d_in[15];
  const float* W2 = (const float*)d_in[16]; const float* b2 = (const float*)d_in[17];
  float* out = (float*)d_out;

  // workspace layout (bytes), all offsets 16B-aligned
  char* ws = (char*)d_ws;
  const size_t SZ_W  = (size_t)D_MODEL * D_MODEL * 2;   // 2 MB
  const size_t SZ_WF = (size_t)D_MODEL * D_FF * 2;      // 8 MB
  const size_t SZ_AC = (size_t)MTOK * D_MODEL * 2;      // 16 MB bf16 act
  const size_t SZ_AF = (size_t)MTOK * D_MODEL * 4;      // 32 MB fp32 act
  const size_t SZ_H  = (size_t)MTOK * D_FF * 2;         // 64 MB

  size_t off = 0;
  bf16* wq_t = (bf16*)(ws + off); off += SZ_W;
  bf16* wk_t = (bf16*)(ws + off); off += SZ_W;
  bf16* wv_t = (bf16*)(ws + off); off += SZ_W;
  bf16* wo_t = (bf16*)(ws + off); off += SZ_W;
  bf16* w1_t = (bf16*)(ws + off); off += SZ_WF;
  bf16* w2_t = (bf16*)(ws + off); off += SZ_WF;
  bf16* xb   = (bf16*)(ws + off); off += SZ_AC;   // ln1 out, later ln2 out
  bf16* qb   = (bf16*)(ws + off); off += SZ_AC;
  bf16* kb_  = (bf16*)(ws + off); off += SZ_AC;
  bf16* vb_  = (bf16*)(ws + off); off += SZ_AC;
  bf16* ab   = (bf16*)(ws + off); off += SZ_AC;   // attention output
  float* xres = (float*)(ws + off); off += SZ_AF;
  bf16* hb   = (bf16*)(ws + off); off += SZ_H;
  (void)ws_size; (void)in_sizes; (void)n_in; (void)out_size;

  const dim3 blk(256);
  const dim3 tblk(32, 8);

  // 1) convert weights fp32 [K][N] -> bf16 [N][K]
  transpose_f32_to_bf16<<<dim3(32, 32), tblk, 0, stream>>>(Wq, wq_t, D_MODEL, D_MODEL);
  transpose_f32_to_bf16<<<dim3(32, 32), tblk, 0, stream>>>(Wk, wk_t, D_MODEL, D_MODEL);
  transpose_f32_to_bf16<<<dim3(32, 32), tblk, 0, stream>>>(Wv, wv_t, D_MODEL, D_MODEL);
  transpose_f32_to_bf16<<<dim3(32, 32), tblk, 0, stream>>>(Wo, wo_t, D_MODEL, D_MODEL);
  transpose_f32_to_bf16<<<dim3(128, 32), tblk, 0, stream>>>(W1, w1_t, D_MODEL, D_FF);
  transpose_f32_to_bf16<<<dim3(32, 128), tblk, 0, stream>>>(W2, w2_t, D_FF, D_MODEL);

  // 2) LN1
  ln_bf16_kernel<<<MTOK, blk, 0, stream>>>(x, ln1_g, ln1_b, xb);

  // 3) QKV projections (mode 0: bf16 out)
  const dim3 gD(D_MODEL / 128, MTOK / 128);
  gemm_bf16_wmma<<<gD, blk, 0, stream>>>(xb, wq_t, bq, nullptr, nullptr, qb,
                                         MTOK, D_MODEL, D_MODEL, 0);
  gemm_bf16_wmma<<<gD, blk, 0, stream>>>(xb, wk_t, bk, nullptr, nullptr, kb_,
                                         MTOK, D_MODEL, D_MODEL, 0);
  gemm_bf16_wmma<<<gD, blk, 0, stream>>>(xb, wv_t, bv, nullptr, nullptr, vb_,
                                         MTOK, D_MODEL, D_MODEL, 0);

  // 4) flash attention
  attn_flash_wmma<<<dim3(SEQ / 128, BATCH * NUM_HEADS), blk, 0, stream>>>(
      qb, kb_, vb_, emask, ab);

  // 5) output projection + residual (mode 2: fp32 out)
  gemm_bf16_wmma<<<gD, blk, 0, stream>>>(ab, wo_t, bo, x, xres, nullptr,
                                         MTOK, D_MODEL, D_MODEL, 2);

  // 6) LN2 (reuse xb)
  ln_bf16_kernel<<<MTOK, blk, 0, stream>>>(xres, ln2_g, ln2_b, xb);

  // 7) FFN1 with ReLU (mode 1)
  gemm_bf16_wmma<<<dim3(D_FF / 128, MTOK / 128), blk, 0, stream>>>(
      xb, w1_t, b1, nullptr, nullptr, hb, MTOK, D_FF, D_MODEL, 1);

  // 8) FFN2 + residual -> d_out (mode 2)
  gemm_bf16_wmma<<<gD, blk, 0, stream>>>(hb, w2_t, b2, xres, out, nullptr,
                                         MTOK, D_MODEL, D_FF, 2);
}
